// CodeQPrevQLSTMModel_Embedding_88759794139611
// MI455X (gfx1250) — matrix operand order repaired and checked by reading, not compiled, not run on hardware
//
#include <hip/hip_runtime.h>
#include <hip/hip_bf16.h>
#include <math.h>

// ---------------------------------------------------------------------------
// Problem constants (from reference)
// ---------------------------------------------------------------------------
#define B_    64
#define Q_    16
#define MAXK_ 32
#define C_    64
#define D_    768
#define H_    512
#define S_    529          // Q*(MAXK+1)+1
#define S1_   530          // S+1 (leading zero row for h_{-1})
#define G_    2048         // 4*H

typedef __attribute__((ext_vector_type(16))) _Float16 v16h;
typedef __attribute__((ext_vector_type(8)))  float    v8f;
typedef __attribute__((__vector_size__(4 * sizeof(int)))) int i32x4_;

// ---------------------------------------------------------------------------
// CDNA5 async global->LDS staging (ASYNCcnt path), with safe fallback.
// Builtin signatures (from hipcc diagnostics):
//   __builtin_amdgcn_global_load_async_to_lds_b128(int4* g, int4* lds, Ii off, Ii cpol)
//   __builtin_amdgcn_global_load_async_to_lds_b32 (int*  g, int*  lds, Ii off, Ii cpol)
// ---------------------------------------------------------------------------
#if defined(__has_builtin)
#  if __has_builtin(__builtin_amdgcn_global_load_async_to_lds_b128)
#    define HAVE_ASYNC128 1
#  else
#    define HAVE_ASYNC128 0
#  endif
#  if __has_builtin(__builtin_amdgcn_global_load_async_to_lds_b32)
#    define HAVE_ASYNC32 1
#  else
#    define HAVE_ASYNC32 0
#  endif
#else
#  define HAVE_ASYNC128 0
#  define HAVE_ASYNC32 0
#endif

__device__ __forceinline__ void stage_b128(const void* g, void* l) {
#if HAVE_ASYNC128
    __builtin_amdgcn_global_load_async_to_lds_b128(
        (__attribute__((address_space(1))) i32x4_*)g,
        (__attribute__((address_space(3))) i32x4_*)l, 0, 0);
#else
    *(uint4*)l = *(const uint4*)g;
#endif
}

__device__ __forceinline__ void stage_b32(const void* g, void* l) {
#if HAVE_ASYNC32
    __builtin_amdgcn_global_load_async_to_lds_b32(
        (__attribute__((address_space(1))) int*)g,
        (__attribute__((address_space(3))) int*)l, 0, 0);
#else
    *(float*)l = *(const float*)g;
#endif
}

__device__ __forceinline__ void wait_async_all() {
#if HAVE_ASYNC128 || HAVE_ASYNC32
#  if __has_builtin(__builtin_amdgcn_s_wait_asynccnt)
    __builtin_amdgcn_s_wait_asynccnt(0);
#  else
    asm volatile("s_wait_asynccnt 0x0" ::: "memory");
#  endif
#endif
}

union FragU { v16h h; uint4 u[2]; };

__device__ __forceinline__ v16h load_frag(const void* p0, const void* p1) {
    FragU f;
    f.u[0] = *(const uint4*)p0;
    f.u[1] = *(const uint4*)p1;
    return f.h;
}

__device__ __forceinline__ float sigf(float x) { return 1.0f / (1.0f + __expf(-x)); }

// ---------------------------------------------------------------------------
// Workspace zeroing (x_f16, hs0, hs1, L, barrier counters)
// ---------------------------------------------------------------------------
__global__ void zero_kernel(uint4* p, size_t n16) {
    size_t i = (size_t)blockIdx.x * blockDim.x + threadIdx.x;
    size_t stride = (size_t)gridDim.x * blockDim.x;
    uint4 z; z.x = 0u; z.y = 0u; z.z = 0u; z.w = 0u;
    for (; i < n16; i += stride) p[i] = z;
}

// ---------------------------------------------------------------------------
// f32 -> f16 weight conversion, bias sum
// ---------------------------------------------------------------------------
__global__ void cvt_f16_kernel(const float* __restrict__ src, _Float16* __restrict__ dst, size_t n) {
    size_t i = (size_t)blockIdx.x * blockDim.x + threadIdx.x;
    size_t stride = (size_t)gridDim.x * blockDim.x;
    for (; i < n; i += stride) dst[i] = (_Float16)src[i];
}

__global__ void bias_sum_kernel(const float* __restrict__ bi, const float* __restrict__ bh,
                                float* __restrict__ out, int n) {
    int i = blockIdx.x * blockDim.x + threadIdx.x;
    if (i < n) out[i] = bi[i] + bh[i];
}

// ---------------------------------------------------------------------------
// build_sequence: scatter questions / code snapshots / text into x (B,S,D) f16,
// compute per-batch length L. One block per batch row.
// ---------------------------------------------------------------------------
__global__ void build_seq_kernel(const float* __restrict__ text,
                                 const float* __restrict__ quest,
                                 const float* __restrict__ code,
                                 const int*   __restrict__ ns,
                                 const int*   __restrict__ qn,
                                 _Float16* __restrict__ x,
                                 int* __restrict__ Lout) {
    const int b = blockIdx.x;
    const int tid = threadIdx.x;
    __shared__ int s_start[Q_];
    __shared__ int s_len[Q_];
    __shared__ int s_L;
    if (tid == 0) {
        int acc = 0;
        int qnb = qn[b];
        for (int q = 0; q < Q_; ++q) {
            int valid = (q < qnb) ? 1 : 0;
            int nsq = valid ? ns[b * Q_ + q] : 0;
            s_len[q] = nsq;
            s_start[q] = acc;
            acc += valid ? (nsq + 1) : 0;
        }
        s_L = acc;
        Lout[b] = acc;
    }
    __syncthreads();
    _Float16* xb = x + (size_t)b * S_ * D_;
    for (int q = 0; q < Q_; ++q) {
        int len = s_len[q];
        if (len <= 0) continue;
        {
            const float* src = quest + ((size_t)b * Q_ + q) * D_;
            _Float16* dst = xb + (size_t)s_start[q] * D_;
            for (int j = tid; j < D_; j += 256) dst[j] = (_Float16)src[j];
        }
        for (int k = 0; k < len; ++k) {
            int srcc = q + k; if (srcc > C_ - 1) srcc = C_ - 1;
            const float* src = code + ((size_t)b * C_ + srcc) * D_;
            _Float16* dst = xb + (size_t)(s_start[q] + 1 + k) * D_;
            for (int j = tid; j < D_; j += 256) dst[j] = (_Float16)src[j];
        }
    }
    {
        const float* src = text + (size_t)b * D_;
        _Float16* dst = xb + (size_t)s_L * D_;
        for (int j = tid; j < D_; j += 256) dst[j] = (_Float16)src[j];
    }
}

// ---------------------------------------------------------------------------
// Input-gate GEMM:  C[M x 2048] = A_f16[M x K] @ Bw_f16[2048 x K]^T + bias
// Tile: 64(M) x 128(N) per workgroup, Ktile=32, double-buffered LDS with
// async global->LDS staging, WMMA f16->f32. remap=1 reads A rows from
// hs layout (B, S+1, H) skipping the leading zero row.
// ---------------------------------------------------------------------------
__global__ void __launch_bounds__(256)
gemm_xg_kernel(const _Float16* __restrict__ A,
               const _Float16* __restrict__ Bw,
               const float* __restrict__ bias,
               float* __restrict__ Cout,
               int K, int remap) {
    __shared__ _Float16 sA[2][64 * 32];    // 2 x 4 KB
    __shared__ _Float16 sB[2][128 * 32];   // 2 x 8 KB  (n-major: [n][k])
    const int tid  = threadIdx.x;
    const int wave = tid >> 5;
    const int lane = tid & 31;
    const int laneM = lane & 15;
    const int kb   = (lane < 16) ? 0 : 8;   // A half-lane K base
    const int ksB  = (lane < 16) ? 0 : 16;  // unused; B also uses kb
    const int m0 = blockIdx.x * 64;
    const int n0 = blockIdx.y * 128;

    // per-thread staging coordinates
    const int arow = tid >> 2;              // 0..63
    const int acol = (tid & 3) * 8;         // 0,8,16,24
    const _Float16* asrc_base;
    {
        int m = m0 + arow;
        if (remap) {
            int bb = m / S_;
            int ss = m - bb * S_;
            asrc_base = A + ((size_t)bb * S1_ + ss + 1) * K + acol;
        } else {
            asrc_base = A + (size_t)m * K + acol;
        }
    }

    v8f zero8 = {0.f, 0.f, 0.f, 0.f, 0.f, 0.f, 0.f, 0.f};
    v8f acc[4];
    acc[0] = zero8; acc[1] = zero8; acc[2] = zero8; acc[3] = zero8;

    const int ktEnd = K >> 5;

    // stage tile kt into LDS buffer `buf`
    auto stage = [&](int kt, int buf) {
        stage_b128(asrc_base + kt * 32, &sA[buf][arow * 32 + acol]);
        for (int i = 0; i < 2; ++i) {
            int idx = tid + i * 256;
            int row = idx >> 2;              // n within tile
            int col = (idx & 3) * 8;
            stage_b128(Bw + (size_t)(n0 + row) * K + kt * 32 + col,
                       &sB[buf][row * 32 + col]);
        }
    };

    stage(0, 0);
    wait_async_all();
    __syncthreads();

    for (int kt = 0; kt < ktEnd; ++kt) {
        const int cur = kt & 1;
        if (kt + 1 < ktEnd) stage(kt + 1, cur ^ 1);   // overlap next tile load

        const _Float16* bp = &sB[cur][(wave * 16 + laneM) * 32 + kb];
        v16h bf = load_frag(bp, bp + 16);
        for (int mi = 0; mi < 4; ++mi) {
            const _Float16* ap = &sA[cur][(mi * 16 + laneM) * 32];
            v16h af = load_frag(ap + kb, ap + 16 + kb);
            acc[mi] = __builtin_amdgcn_wmma_f32_16x16x32_f16(
                false, af, false, bf, (short)0, acc[mi], false, false);
        }
        wait_async_all();
        __syncthreads();
    }

    // store C + bias  (f32 C/D layout: VGPR r -> M = r + 8*(lane>=16))
    const int n_g = n0 + wave * 16 + laneM;
    const float bv = bias[n_g];
    const int mrow = (lane & 16) ? 8 : 0;
    for (int mi = 0; mi < 4; ++mi) {
        for (int r = 0; r < 8; ++r) {
            int m = m0 + mi * 16 + mrow + r;
            Cout[(size_t)m * G_ + n_g] = acc[mi][r] + bv;
        }
    }
}

// ---------------------------------------------------------------------------
// Persistent LSTM scan: 16 workgroups, each owns 32 h-columns (=128 gate cols).
// LDS (dynamic, 264 KB):
//   sW  [128][512] f16  128 KB  w_hh slice, resident for all 529 steps
//   sH  [ 64][512] f16   64 KB  h_{t-1}, staged once per step (async b128)
//   sG  2x[64][128] f32  64 KB  gate accumulators, xg double-buffered (async b32)
//   sC  [ 64][ 32] f32    8 KB  cell state
// Per step: WMMA-accumulate h_{t-1} @ w_hh^T into sG -> gates -> write h_t
// (f16) -> grid barrier (monotonic atomic + s_sleep spin).
// ---------------------------------------------------------------------------
__global__ void __launch_bounds__(256)
lstm_scan_kernel(const float* __restrict__ xg,
                 const _Float16* __restrict__ whh,
                 _Float16* __restrict__ hs,
                 unsigned* __restrict__ counter) {
    extern __shared__ char smem[];
    _Float16* sW  = (_Float16*)smem;                               // 131072 B
    _Float16* sH  = (_Float16*)(smem + 131072);                    //  65536 B
    float*    sG0 = (float*)(smem + 131072 + 65536);               //  32768 B
    float*    sG1 = (float*)(smem + 131072 + 65536 + 32768);       //  32768 B
    float*    sC  = (float*)(smem + 131072 + 65536 + 65536);       //   8192 B

    const int tid  = threadIdx.x;
    const int wg   = blockIdx.x;        // 0..15 -> h-cols [wg*32, wg*32+32)
    const int wave = tid >> 5;
    const int lane = tid & 31;
    const int laneM = lane & 15;
    const int kb   = (lane < 16) ? 0 : 8;
    const int ksB  = (lane < 16) ? 0 : 16;
    const unsigned nwg = gridDim.x;

    // Load w_hh slice once: local gate row r -> global col (r/32)*512 + wg*32 + (r%32)
    for (int i = 0; i < 32; ++i) {
        int e = (tid + i * 256) * 8;    // f16 element index into [128][512]
        int r = e >> 9;
        int k = e & 511;
        int gc = ((r >> 5) << 9) + wg * 32 + (r & 31);
        stage_b128(whh + (size_t)gc * H_ + k, sW + r * 512 + k);
    }
    for (int i = tid; i < 64 * 32; i += 256) sC[i] = 0.0f;

    // stage xg gate slice for step t into dst (32 b32 per thread, coalesced)
    auto stage_xg = [&](int t, float* dst) {
        for (int i = 0; i < 32; ++i) {
            int idx = tid + i * 256;            // = m*128 + r
            int m = idx >> 7;
            int r = idx & 127;
            int gcol = ((r >> 5) << 9) + wg * 32 + (r & 31);
            stage_b32(xg + ((size_t)m * S_ + t) * G_ + gcol, dst + idx);
        }
    };

    stage_xg(0, sG0);                   // prologue: gates for t=0
    wait_async_all();
    __syncthreads();

    v8f zero8 = {0.f, 0.f, 0.f, 0.f, 0.f, 0.f, 0.f, 0.f};

    for (int t = 0; t < S_; ++t) {
        float* sG  = (t & 1) ? sG1 : sG0;
        float* sGn = (t & 1) ? sG0 : sG1;

        // phase 0: stage h_{t-1} (async b128) + next step's xg (async b32)
        for (int i = 0; i < 16; ++i) {
            int e = (tid + i * 256) * 8;        // f16 index into [64][512]
            int m = e >> 9;
            int k = e & 511;
            stage_b128(hs + ((size_t)m * S1_ + t) * H_ + k, sH + e);
        }
        if (t + 1 < S_) stage_xg(t + 1, sGn);
        if (t + 1 < S_)
            __builtin_prefetch(xg + ((size_t)(tid & 63) * S_ + (t + 1)) * G_ + wg * 32, 0, 1);
        wait_async_all();
        __syncthreads();

        // phase 1: recurrent GEMM h_{t-1}[64x512] @ w_hh_slice^T[512x128]
        v8f acc[4];
        acc[0] = zero8; acc[1] = zero8; acc[2] = zero8; acc[3] = zero8;
        const _Float16* sWrow = sW + (wave * 16 + laneM) * 512;
        for (int kt = 0; kt < 16; ++kt) {
            int ks = kb + kt * 32;
            v16h bf = load_frag(sWrow + ks, sWrow + ks + 16);
            for (int mi = 0; mi < 4; ++mi) {
                const _Float16* ap = sH + (mi * 16 + laneM) * 512 + kt * 32 + kb;
                v16h af = load_frag(ap, ap + 16);
                acc[mi] = __builtin_amdgcn_wmma_f32_16x16x32_f16(
                    false, af, false, bf, (short)0, acc[mi], false, false);
            }
        }
        {
            int r = wave * 16 + laneM;
            int mrow = (lane & 16) ? 8 : 0;
            for (int mi = 0; mi < 4; ++mi)
                for (int rr = 0; rr < 8; ++rr) {
                    int m = mi * 16 + mrow + rr;
                    sG[m * 128 + r] += acc[mi][rr];
                }
        }
        __syncthreads();

        // phase 2: gates -> c, h ; write h_t (f16) for next step / next layer
        for (int i = 0; i < 8; ++i) {
            int idx = tid + i * 256;            // 0..2047
            int b = idx >> 5;
            int j = idx & 31;
            float iv = sG[b * 128 + j];
            float fv = sG[b * 128 + 32 + j];
            float gv = sG[b * 128 + 64 + j];
            float ov = sG[b * 128 + 96 + j];
            float c = sigf(fv) * sC[b * 32 + j] + sigf(iv) * tanhf(gv);
            sC[b * 32 + j] = c;
            float h = sigf(ov) * tanhf(c);
            hs[((size_t)b * S1_ + t + 1) * H_ + wg * 32 + j] = (_Float16)h;
        }
        __threadfence();
        __syncthreads();

        // phase 3: grid-wide barrier (monotonic counter, re-zeroed every launch)
        if (tid == 0) {
            __hip_atomic_fetch_add(counter, 1u, __ATOMIC_RELEASE, __HIP_MEMORY_SCOPE_AGENT);
            unsigned target = nwg * (unsigned)(t + 1);
            while (__hip_atomic_load(counter, __ATOMIC_ACQUIRE, __HIP_MEMORY_SCOPE_AGENT) < target) {
                __builtin_amdgcn_s_sleep(1);
            }
        }
        __syncthreads();
    }
}

// ---------------------------------------------------------------------------
// Gather last h + FC:  out[b] = hs1[b, L[b]] . fc_w + fc_b
// ---------------------------------------------------------------------------
__global__ void final_kernel(const _Float16* __restrict__ hs1,
                             const int* __restrict__ L,
                             const float* __restrict__ fcw,
                             const float* __restrict__ fcb,
                             float* __restrict__ out) {
    __shared__ float red[256];
    const int b = blockIdx.x;
    const int tid = threadIdx.x;
    const _Float16* h = hs1 + ((size_t)b * S1_ + L[b] + 1) * H_;
    float s = 0.0f;
    for (int j = tid; j < H_; j += 256) s += (float)h[j] * fcw[j];
    red[tid] = s;
    __syncthreads();
    for (int o = 128; o > 0; o >>= 1) {
        if (tid < o) red[tid] += red[tid + o];
        __syncthreads();
    }
    if (tid == 0) out[b] = red[0] + fcb[0];
}

// ---------------------------------------------------------------------------
// Host side
// ---------------------------------------------------------------------------
extern "C" void kernel_launch(void* const* d_in, const int* in_sizes, int n_in,
                              void* d_out, int out_size, void* d_ws, size_t ws_size,
                              hipStream_t stream) {
    const float* text  = (const float*)d_in[0];
    const float* quest = (const float*)d_in[1];
    const float* code  = (const float*)d_in[2];
    const int*   ns    = (const int*)d_in[3];
    const int*   qn    = (const int*)d_in[4];
    const float* wih0  = (const float*)d_in[5];
    const float* whh0  = (const float*)d_in[6];
    const float* bih0  = (const float*)d_in[7];
    const float* bhh0  = (const float*)d_in[8];
    const float* wih1  = (const float*)d_in[9];
    const float* whh1  = (const float*)d_in[10];
    const float* bih1  = (const float*)d_in[11];
    const float* bhh1  = (const float*)d_in[12];
    const float* fcw   = (const float*)d_in[13];
    const float* fcb   = (const float*)d_in[14];
    float* out = (float*)d_out;

    char* ws = (char*)d_ws;
    size_t off = 0;
    auto alloc = [&](size_t bytes) { size_t o = off; off = (off + bytes + 255) & ~(size_t)255; return o; };

    const size_t o_xf16 = alloc((size_t)B_ * S_ * D_ * 2);     // 52.0 MB
    const size_t o_hs0  = alloc((size_t)B_ * S1_ * H_ * 2);    // 34.7 MB
    const size_t o_hs1  = alloc((size_t)B_ * S1_ * H_ * 2);    // 34.7 MB
    const size_t o_L    = alloc((size_t)B_ * 4);
    const size_t o_ctr  = alloc(256);
    const size_t zero_end = off;                               // zero everything above
    const size_t o_xg   = alloc((size_t)B_ * S_ * G_ * 4);     // 277 MB (reused by both layers)
    const size_t o_wih0 = alloc((size_t)G_ * D_ * 2);
    const size_t o_whh0 = alloc((size_t)G_ * H_ * 2);
    const size_t o_wih1 = alloc((size_t)G_ * H_ * 2);
    const size_t o_whh1 = alloc((size_t)G_ * H_ * 2);
    const size_t o_b0   = alloc((size_t)G_ * 4);
    const size_t o_b1   = alloc((size_t)G_ * 4);
    (void)ws_size; (void)n_in; (void)in_sizes; (void)out_size;

    _Float16* xf16 = (_Float16*)(ws + o_xf16);
    _Float16* hs0  = (_Float16*)(ws + o_hs0);
    _Float16* hs1  = (_Float16*)(ws + o_hs1);
    int*      Lb   = (int*)(ws + o_L);
    unsigned* ctr0 = (unsigned*)(ws + o_ctr);
    unsigned* ctr1 = (unsigned*)(ws + o_ctr + 64);
    float*    xg   = (float*)(ws + o_xg);
    _Float16* fw_ih0 = (_Float16*)(ws + o_wih0);
    _Float16* fw_hh0 = (_Float16*)(ws + o_whh0);
    _Float16* fw_ih1 = (_Float16*)(ws + o_wih1);
    _Float16* fw_hh1 = (_Float16*)(ws + o_whh1);
    float*    bias0  = (float*)(ws + o_b0);
    float*    bias1  = (float*)(ws + o_b1);

    // 1) zero x_f16 / hs buffers / L / barrier counters (every call: deterministic)
    zero_kernel<<<4096, 256, 0, stream>>>((uint4*)ws, zero_end / 16);

    // 2) build sequence + lengths
    build_seq_kernel<<<B_, 256, 0, stream>>>(text, quest, code, ns, qn, xf16, Lb);

    // 3) weight conversion + bias sums
    cvt_f16_kernel<<<512, 256, 0, stream>>>(wih0, fw_ih0, (size_t)G_ * D_);
    cvt_f16_kernel<<<512, 256, 0, stream>>>(whh0, fw_hh0, (size_t)G_ * H_);
    cvt_f16_kernel<<<512, 256, 0, stream>>>(wih1, fw_ih1, (size_t)G_ * H_);
    cvt_f16_kernel<<<512, 256, 0, stream>>>(whh1, fw_hh1, (size_t)G_ * H_);
    bias_sum_kernel<<<8, 256, 0, stream>>>(bih0, bhh0, bias0, G_);
    bias_sum_kernel<<<8, 256, 0, stream>>>(bih1, bhh1, bias1, G_);

    // M = B*S = 33856 -> 529 tiles of 64 ; N = 2048 -> 16 tiles of 128
    dim3 gemm_grid((B_ * S_) / 64, G_ / 128, 1);

    // 4) layer-0 input gates: xg = x_f16 @ w_ih0^T + (b_ih0+b_hh0)
    gemm_xg_kernel<<<gemm_grid, 256, 0, stream>>>(xf16, fw_ih0, bias0, xg, D_, 0);

    // 5) layer-0 recurrence (persistent, LDS-resident weights, grid barrier)
    const size_t scan_lds = 131072 + 65536 + 32768 + 32768 + 8192; // 270336 B
    lstm_scan_kernel<<<16, 256, scan_lds, stream>>>(xg, fw_hh0, hs0, ctr0);

    // 6) layer-1 input gates from hs0 (remapped (B,S+1,H) rows)
    gemm_xg_kernel<<<gemm_grid, 256, 0, stream>>>(hs0, fw_ih1, bias1, xg, H_, 1);

    // 7) layer-1 recurrence
    lstm_scan_kernel<<<16, 256, scan_lds, stream>>>(xg, fw_hh1, hs1, ctr1);

    // 8) gather + FC
    final_kernel<<<B_, 256, 0, stream>>>(hs1, Lb, fcw, fcb, out);
}
